// VectorQuantizer_22600117912221
// MI455X (gfx1250) — compile-verified, matrix operand order checked by reading, not run on hardware
//
#include <hip/hip_runtime.h>
#include <hip/hip_bf16.h>

// ---------------------------------------------------------------------------
// VQ quantizer for MI455X (gfx1250, wave32).
//   score[n][e] = ||e||^2 - 2 * dot(x_n, emb_e)      (x-norm is row-constant)
// GEMM via V_WMMA_F32_16X16X32_BF16 with a hi/lo bf16 split ("bf16x3"):
//   x = xh + xl, e = eh + el  (xh=bf16(x), xl=bf16(x-xh))
//   dot ~= xh.eh + xh.el + xl.eh   (missing xl.el ~ 2^-18 rel, < f32 noise)
// Codebook pre-split into bf16 hi/lo planes once; argmin kernel stages tiles
// with double-buffered GLOBAL_LOAD_ASYNC_TO_LDS_B128 (ASYNCcnt). Argmin kept
// as per-lane running minima; single cross-lane reduction at the end.
// ---------------------------------------------------------------------------

typedef float  v8f   __attribute__((ext_vector_type(8)));
typedef int    v4i   __attribute__((ext_vector_type(4)));
typedef __bf16 v4bf  __attribute__((ext_vector_type(4)));
typedef __bf16 v8bf  __attribute__((ext_vector_type(8)));
typedef __bf16 v16bf __attribute__((ext_vector_type(16)));

#define NROWS 65536
#define NEMB  1024
#define DIM   256
#define LDSW  264   // bf16 row stride: 132 dwords, %64==4 -> conflict-free b128

#if defined(__gfx1250__) && __has_builtin(__builtin_amdgcn_global_load_async_to_lds_b128)
#define HAS_ASYNC_LDS 1
#else
#define HAS_ASYNC_LDS 0
#endif

#if HAS_ASYNC_LDS
# if __has_builtin(__builtin_amdgcn_s_wait_asynccnt)
#  define WAIT_ASYNCCNT(n) __builtin_amdgcn_s_wait_asynccnt(n)
# else
#  define WAIT_ASYNCCNT(n) asm volatile("s_wait_asynccnt %0" ::"i"(n) : "memory")
# endif
#else
#define WAIT_ASYNCCNT(n) ((void)0)
#endif

// ---------------------------------------------------------------------------
// Pass 0: per codebook row, compute ||e||^2 (exact f32) and split the row
// into bf16 hi/lo planes (written once; reused by all 512 argmin WGs).
// One wave per embedding row.
// ---------------------------------------------------------------------------
__global__ __launch_bounds__(256) void vq_prep_kernel(
    const float* __restrict__ emb, float* __restrict__ enorm,
    __bf16* __restrict__ embH, __bf16* __restrict__ embL) {
  const int e    = blockIdx.x * 8 + (threadIdx.x >> 5);
  const int lane = threadIdx.x & 31;
  const float4 a = *(const float4*)(emb + e * DIM + lane * 4);
  const float4 b = *(const float4*)(emb + e * DIM + 128 + lane * 4);
  const float f[8] = {a.x, a.y, a.z, a.w, b.x, b.y, b.z, b.w};
  v4bf h0, h1, l0, l1;
  float s = 0.0f;
  #pragma unroll
  for (int j = 0; j < 4; ++j) {
    s += f[j] * f[j] + f[4 + j] * f[4 + j];
    const __bf16 ha = (__bf16)f[j];
    const __bf16 hb = (__bf16)f[4 + j];
    h0[j] = ha; l0[j] = (__bf16)(f[j] - (float)ha);
    h1[j] = hb; l1[j] = (__bf16)(f[4 + j] - (float)hb);
  }
  *(v4bf*)&embH[e * DIM + lane * 4]       = h0;
  *(v4bf*)&embH[e * DIM + 128 + lane * 4] = h1;
  *(v4bf*)&embL[e * DIM + lane * 4]       = l0;
  *(v4bf*)&embL[e * DIM + 128 + lane * 4] = l1;
  #pragma unroll
  for (int off = 16; off > 0; off >>= 1) s += __shfl_xor(s, off, 32);
  if (lane == 0) enorm[e] = s;
}

// ---------------------------------------------------------------------------
// Stage one 16x256 bf16 plane tile into (padded) LDS: 2 b128 chunks/thread.
// Async path: fire-and-forget via ASYNCcnt; fallback: plain copy.
// ---------------------------------------------------------------------------
__device__ __forceinline__ void stage_tile(const __bf16* __restrict__ gplane,
                                           __bf16* lplane, int t, int tid) {
  #pragma unroll
  for (int s2 = 0; s2 < 2; ++s2) {
    const int cid = tid + s2 * 256;        // 0..511 chunk id (16B chunks)
    const int r   = cid >> 5;              // 32 chunks per 256-elem row
    const int cc  = (cid & 31) * 8;        // bf16 element offset in row
    const __bf16* g = gplane + ((size_t)t * 16 + r) * DIM + cc;
    __bf16* l = lplane + r * LDSW + cc;
#if HAS_ASYNC_LDS
    __builtin_amdgcn_global_load_async_to_lds_b128((v4i*)g, (v4i*)l, 0, 0);
#else
    *(v8bf*)l = *(const v8bf*)g;
#endif
  }
}

// ---------------------------------------------------------------------------
// Pass 1: fused bf16x3 GEMM + row argmin.
// WG = 256 threads = 8 waves; each wave owns 16 rows of x.
// 16-bit A 16x32 layout (ISA 7.12.2): lane L -> row M=L&15; the 16 bf16 in a
// fragment are two 8-element K runs: K = 32*kk + 8*(L>>4) + {0..7} and +16.
// B (32x16) mirrors with N = L&15. C: VGPR i -> M = i + 8*(L>>4), N = L&15.
// Tile pipeline: issue t+1 async into alt buffer, s_wait_asynccnt<=4 (in-order
// completion => tile t landed), barrier, 24 chained WMMAs, barrier.
// Argmin: lane tracks running min over its own column subset {16t + lane&15};
// one cross-lane (16-lane) reduction with first-index tie-break at the end.
// ---------------------------------------------------------------------------
__global__ __launch_bounds__(256) void vq_argmin_kernel(
    const float* __restrict__ x, const __bf16* __restrict__ embH,
    const __bf16* __restrict__ embL, const float* __restrict__ enorm,
    int* __restrict__ inds, float* __restrict__ inds_f) {
  __shared__ __align__(16) __bf16 ldsH[2][16 * LDSW];
  __shared__ __align__(16) __bf16 ldsL[2][16 * LDSW];

  const int tid  = threadIdx.x;
  const int lane = tid & 31;
  const int wave = tid >> 5;
  const int half = lane >> 4;
  const int l15  = lane & 15;
  const int rowBase = blockIdx.x * 128 + wave * 16;
  const int m = rowBase + l15;

  // Build A fragments once (hi+lo), reused across all 64 codebook tiles.
  v16bf ah[8], al[8];
  {
    const float* xr = x + (size_t)m * DIM;
    #pragma unroll
    for (int kk = 0; kk < 8; ++kk) {
      const int k0 = kk * 32 + 8 * half;
      const float4 u0 = *(const float4*)(xr + k0);
      const float4 u1 = *(const float4*)(xr + k0 + 4);
      const float4 u2 = *(const float4*)(xr + k0 + 16);
      const float4 u3 = *(const float4*)(xr + k0 + 20);
      const float f[16] = {u0.x, u0.y, u0.z, u0.w, u1.x, u1.y, u1.z, u1.w,
                           u2.x, u2.y, u2.z, u2.w, u3.x, u3.y, u3.z, u3.w};
      #pragma unroll
      for (int j = 0; j < 16; ++j) {
        const __bf16 h = (__bf16)f[j];
        ah[kk][j] = h;
        al[kk][j] = (__bf16)(f[j] - (float)h);
      }
    }
  }

  float rmin[8];
  int   rarg[8];
  #pragma unroll
  for (int i = 0; i < 8; ++i) { rmin[i] = 3.0e38f; rarg[i] = 0; }

  // Prologue: stage tile 0 into buffer 0.
  stage_tile(embH, &ldsH[0][0], 0, tid);
  stage_tile(embL, &ldsL[0][0], 0, tid);

  for (int t = 0; t < NEMB / 16; ++t) {
    const int cur = t & 1;
    if (t < NEMB / 16 - 1) {
      // Alt buffer was last read at tile t-1; barrier [E] below made it safe.
      stage_tile(embH, &ldsH[cur ^ 1][0], t + 1, tid);
      stage_tile(embL, &ldsL[cur ^ 1][0], t + 1, tid);
      WAIT_ASYNCCNT(4);   // <=4 outstanding => tile t's 4 loads completed
    } else {
      WAIT_ASYNCCNT(0);
    }
    __syncthreads();      // [C] every wave's tile-t loads are in LDS

    const __bf16* bH = &ldsH[cur][0];
    const __bf16* bL = &ldsL[cur][0];
    v8f acc = {};
    #pragma unroll
    for (int kk = 0; kk < 8; ++kk) {
      const int o = l15 * LDSW + kk * 32 + 8 * half;
      const v8bf bh0 = *(const v8bf*)&bH[o];
      const v8bf bh1 = *(const v8bf*)&bH[o + 16];
      const v8bf bl0 = *(const v8bf*)&bL[o];
      const v8bf bl1 = *(const v8bf*)&bL[o + 16];
      const v16bf bh = __builtin_shufflevector(
          bh0, bh1, 0, 1, 2, 3, 4, 5, 6, 7, 8, 9, 10, 11, 12, 13, 14, 15);
      const v16bf bl = __builtin_shufflevector(
          bl0, bl1, 0, 1, 2, 3, 4, 5, 6, 7, 8, 9, 10, 11, 12, 13, 14, 15);
      acc = __builtin_amdgcn_wmma_f32_16x16x32_bf16(
          false, ah[kk], false, bh, (short)0, acc, false, false);
      acc = __builtin_amdgcn_wmma_f32_16x16x32_bf16(
          false, ah[kk], false, bl, (short)0, acc, false, false);
      acc = __builtin_amdgcn_wmma_f32_16x16x32_bf16(
          false, al[kk], false, bh, (short)0, acc, false, false);
    }

    // Per-lane running argmin on v = 0.5*||e||^2 - dot (exact *0.5 scaling:
    // same ordering and ties as ||e||^2 - 2*dot). Strict < keeps the
    // earliest (lowest) column within this lane's ascending column stream.
    const float en2 = 0.5f * enorm[t * 16 + l15];
    const int   col = t * 16 + l15;
    #pragma unroll
    for (int i = 0; i < 8; ++i) {
      const float v = en2 - acc[i];
      if (v < rmin[i]) { rmin[i] = v; rarg[i] = col; }
    }
    __syncthreads();      // [E] all reads of buf[cur] done before reissue
  }

  // Single cross-lane min+argmin per row across the 16 lanes of each C-half
  // (xor masks < 16 stay inside the half). Ties -> lowest column index.
  #pragma unroll
  for (int i = 0; i < 8; ++i) {
    float v = rmin[i];
    int   c = rarg[i];
    #pragma unroll
    for (int off = 1; off < 16; off <<= 1) {
      const float ov = __shfl_xor(v, off, 32);
      const int   oc = __shfl_xor(c, off, 32);
      if (ov < v || (ov == v && oc < c)) { v = ov; c = oc; }
    }
    rarg[i] = c;
  }

  // Lanes 0 and 16 publish the 8 rows of their half.
  if (l15 == 0) {
    #pragma unroll
    for (int i = 0; i < 8; ++i) {
      const int r = rowBase + 8 * half + i;
      inds[r]   = rarg[i];
      inds_f[r] = (float)rarg[i];
    }
  }
}

// ---------------------------------------------------------------------------
// Pass 2: gather x_q = emb[idx] (exact f32) and per-block sum of (x_q - x)^2.
// 64 threads per row, float4 each; deterministic block-tree reduction.
// ---------------------------------------------------------------------------
__global__ __launch_bounds__(256) void vq_gather_kernel(
    const float* __restrict__ x, const float* __restrict__ emb,
    const int* __restrict__ inds, float* __restrict__ xq,
    float* __restrict__ partials) {
  __shared__ float red[256];
  const int tid = blockIdx.x * 256 + threadIdx.x;
  const int r   = tid >> 6;
  const int c   = (tid & 63) * 4;
  const int idx = inds[r];
  const float4 e4 = *(const float4*)(emb + idx * DIM + c);
  const float4 x4 = *(const float4*)(x + (size_t)r * DIM + c);
  *(float4*)(xq + (size_t)r * DIM + c) = e4;
  const float dx = e4.x - x4.x, dy = e4.y - x4.y;
  const float dz = e4.z - x4.z, dw = e4.w - x4.w;
  red[threadIdx.x] = dx * dx + dy * dy + dz * dz + dw * dw;
  __syncthreads();
  #pragma unroll
  for (int o = 128; o > 0; o >>= 1) {
    if (threadIdx.x < o) red[threadIdx.x] += red[threadIdx.x + o];
    __syncthreads();
  }
  if (threadIdx.x == 0) partials[blockIdx.x] = red[0];
}

// ---------------------------------------------------------------------------
// Pass 3: final deterministic reduction -> loss = 1.25 * mean((xq - x)^2)
// (codebook_loss + COMMITMENT_COST * commitment_loss, identical values).
// ---------------------------------------------------------------------------
__global__ __launch_bounds__(256) void vq_loss_kernel(
    const float* __restrict__ partials, int n, float* __restrict__ loss_out) {
  __shared__ float red[256];
  float s = 0.0f;
  for (int i = threadIdx.x; i < n; i += 256) s += partials[i];
  red[threadIdx.x] = s;
  __syncthreads();
  #pragma unroll
  for (int o = 128; o > 0; o >>= 1) {
    if (threadIdx.x < o) red[threadIdx.x] += red[threadIdx.x + o];
    __syncthreads();
  }
  if (threadIdx.x == 0)
    *loss_out = red[0] * (1.25f / ((float)NROWS * (float)DIM));
}

// ---------------------------------------------------------------------------
extern "C" void kernel_launch(void* const* d_in, const int* in_sizes, int n_in,
                              void* d_out, int out_size, void* d_ws, size_t ws_size,
                              hipStream_t stream) {
  (void)in_sizes; (void)n_in; (void)out_size; (void)ws_size;

  const float* x   = (const float*)d_in[0];   // [NROWS, DIM] f32
  const float* emb = (const float*)d_in[1];   // [NEMB, DIM]  f32

  // d_out: x_q [NROWS*DIM] | loss [1] | indices-as-float [NROWS]
  float* out    = (float*)d_out;
  float* xq     = out;
  float* loss   = out + (size_t)NROWS * DIM;
  float* inds_f = loss + 1;

  // workspace layout (16B-aligned offsets):
  //   enorm 4KB | inds 256KB | partials 64KB | embH 512KB | embL 512KB
  char*   ws       = (char*)d_ws;
  float*  enorm    = (float*)ws;
  int*    inds     = (int*)(ws + 4096);
  float*  partials = (float*)(ws + 266240);
  __bf16* embH     = (__bf16*)(ws + 331776);
  __bf16* embL     = (__bf16*)(ws + 856064);

  const int nPart = (NROWS * 64) / 256;   // 16384 gather blocks

  vq_prep_kernel  <<<NEMB / 8,    256, 0, stream>>>(emb, enorm, embH, embL);
  vq_argmin_kernel<<<NROWS / 128, 256, 0, stream>>>(x, embH, embL, enorm, inds, inds_f);
  vq_gather_kernel<<<nPart,       256, 0, stream>>>(x, emb, inds, xq, partials);
  vq_loss_kernel  <<<1,           256, 0, stream>>>(partials, nPart, loss);
}
